// DiffPool_6373731467801
// MI455X (gfx1250) — matrix operand order repaired and checked by reading, not compiled, not run on hardware
//
#include <hip/hip_runtime.h>

#define NN 20000
#define EE 320000
#define CC 64
#define FF 128

typedef __attribute__((ext_vector_type(16))) _Float16 v16h;
typedef __attribute__((ext_vector_type(8)))  float    v8f;

// Async global->LDS path (gfx1250) if this toolchain declares the builtins.
#if defined(__has_builtin)
#  if __has_builtin(__builtin_amdgcn_global_load_async_to_lds_b32) && \
      __has_builtin(__builtin_amdgcn_s_wait_asynccnt)
#    define USE_ASYNC_LDS 1
#  endif
#endif
#ifndef USE_ASYNC_LDS
#  define USE_ASYNC_LDS 0
#endif

#if USE_ASYNC_LDS
typedef __attribute__((address_space(1))) int as1_int;
typedef __attribute__((address_space(3))) int as3_int;
#endif

// ---------------------------------------------------------------------------
// Fast WMMA GEMM, Nc fixed at 64:  C[M,64] = A[M,K](ld=lda) * B[K,64] (+bias)(+C)(relu)
// Block = 128 threads (4 waves), macro-tile 64(M) x 64(N).
// Per wave: one 16-row A fragment -> 4 x v_wmma_f32_16x16x32_f16 per K-step.
// B panel [32 x 64] staged K-transposed in LDS (Bs[col*32+k]) so each lane's
// B fragment is two runs of 8 contiguous floats.  K must be a multiple of 32.
// Fragment layouts per CDNA5 ISA 7.12.2 (wave32).
// ---------------------------------------------------------------------------
__global__ __launch_bounds__(128)
void wmma_gemm_fast_kernel(const float* __restrict__ A, int lda,
                           const float* __restrict__ B,
                           float* __restrict__ C,
                           int M, int K,
                           const float* __restrict__ bias,
                           int accumulate, int do_relu)
{
    __shared__ __attribute__((aligned(16))) float Bs[64 * 32];  // [col][k]

    const int tid      = threadIdx.x;
    const int wave     = tid >> 5;
    const int lane     = tid & 31;
    const int half_sel = lane >> 4;
    const int lrow     = lane & 15;
    const int tileM    = blockIdx.x * 64 + wave * 16;

    v8f acc[4] = {};
#pragma unroll
    for (int n = 0; n < 4; ++n) {
        const int col = n * 16 + lrow;
        float bv = bias ? bias[col] : 0.0f;
#pragma unroll
        for (int v = 0; v < 8; ++v) {
            int row = tileM + v + 8 * half_sel;
            float c0 = 0.0f;
            if (accumulate && row < M) c0 = C[(size_t)row * 64 + col];
            acc[n][v] = c0 + bv;
        }
    }

    const int arow  = tileM + lrow;
    const int arowc = (arow < M) ? arow : (M - 1);   // clamp loads; stores guarded

    // B panel staging map: thread t covers k = t>>2, cols (t&3)*16 .. +15
    const int sk  = tid >> 2;
    const int sc0 = (tid & 3) * 16;

    for (int k0 = 0; k0 < K; k0 += 32) {
        __syncthreads();   // protect Bs from previous iteration's readers
        {
            const float* brow = B + (size_t)(k0 + sk) * 64 + sc0;
#if USE_ASYNC_LDS
#pragma unroll
            for (int j = 0; j < 16; ++j) {
                __builtin_amdgcn_global_load_async_to_lds_b32(
                    (as1_int*)(brow + j),
                    (as3_int*)(&Bs[(sc0 + j) * 32 + sk]),
                    0, 0);
            }
            __builtin_amdgcn_s_wait_asynccnt(0);
#else
#pragma unroll
            for (int j = 0; j < 16; ++j) {
                Bs[(sc0 + j) * 32 + sk] = brow[j];
            }
#endif
        }
        __syncthreads();

        // A fragment: rows contiguous in K -> 4 x b128 loads + pack to f16
        const float* ap = A + (size_t)arowc * lda + k0 + 8 * half_sel;
        const float4 a0 = *(const float4*)(ap + 0);
        const float4 a1 = *(const float4*)(ap + 4);
        const float4 a2 = *(const float4*)(ap + 16);
        const float4 a3 = *(const float4*)(ap + 20);
        v16h afrag = {};
        afrag[0]  = (_Float16)a0.x;  afrag[1]  = (_Float16)a0.y;
        afrag[2]  = (_Float16)a0.z;  afrag[3]  = (_Float16)a0.w;
        afrag[4]  = (_Float16)a1.x;  afrag[5]  = (_Float16)a1.y;
        afrag[6]  = (_Float16)a1.z;  afrag[7]  = (_Float16)a1.w;
        afrag[8]  = (_Float16)a2.x;  afrag[9]  = (_Float16)a2.y;
        afrag[10] = (_Float16)a2.z;  afrag[11] = (_Float16)a2.w;
        afrag[12] = (_Float16)a3.x;  afrag[13] = (_Float16)a3.y;
        afrag[14] = (_Float16)a3.z;  afrag[15] = (_Float16)a3.w;

#pragma unroll
        for (int n = 0; n < 4; ++n) {
            const float* bp = &Bs[(n * 16 + lrow) * 32 + 8 * half_sel];
            v16h bfrag = {};
#pragma unroll
            for (int e = 0; e < 16; ++e) bfrag[e] = (_Float16)bp[(e & 7) + ((e >> 3) << 4)];
            acc[n] = __builtin_amdgcn_wmma_f32_16x16x32_f16(
                false, afrag, false, bfrag, (short)0, acc[n], false, false);
        }
    }

#pragma unroll
    for (int n = 0; n < 4; ++n) {
        const int col = n * 16 + lrow;
#pragma unroll
        for (int v = 0; v < 8; ++v) {
            int row = tileM + v + 8 * half_sel;
            if (row < M) {
                float val = acc[n][v];
                if (do_relu) val = fmaxf(val, 0.0f);
                C[(size_t)row * 64 + col] = val;
            }
        }
    }
}

// ---------------------------------------------------------------------------
// Split-K WMMA GEMM for C[64,64] = P^T[64,K] * B[K,64], P is [K,64] row-major.
// grid = (4, 4, numKChunks); partials combined via global_atomic_add_f32.
// C must be zero-initialized.
// ---------------------------------------------------------------------------
__global__ __launch_bounds__(32)
void wmma_gemm_tA_kernel(const float* __restrict__ P,
                         const float* __restrict__ B,
                         float* __restrict__ C,
                         int K, int kChunk)
{
    const int lane     = threadIdx.x & 31;
    const int half_sel = lane >> 4;
    const int lrow     = lane & 15;
    const int tileM    = blockIdx.x * 16;
    const int tileN    = blockIdx.y * 16;
    const int col      = tileN + lrow;
    const int arow     = tileM + lrow;

    int k_begin = blockIdx.z * kChunk;
    int k_end   = k_begin + kChunk;
    if (k_end > K) k_end = K;

    v8f acc = {};
    for (int k0 = k_begin; k0 < k_end; k0 += 32) {
        v16h afrag = {};
        v16h bfrag = {};
        const int ka = k0 + 8 * half_sel;
#pragma unroll
        for (int e = 0; e < 8; ++e)
            afrag[e] = (_Float16)P[(size_t)(ka + e) * 64 + arow];        // A=P^T
#pragma unroll
        for (int e = 0; e < 8; ++e)
            afrag[8 + e] = (_Float16)P[(size_t)(ka + 16 + e) * 64 + arow];
        const int kb = k0 + 8 * half_sel;
#pragma unroll
        for (int e = 0; e < 16; ++e)
            bfrag[e] = (_Float16)B[(size_t)(kb + (e & 7) + ((e >> 3) << 4)) * 64 + col];
        acc = __builtin_amdgcn_wmma_f32_16x16x32_f16(
            false, afrag, false, bfrag, (short)0, acc, false, false);
    }

#pragma unroll
    for (int v = 0; v < 8; ++v) {
        int row = tileM + v + 8 * half_sel;
        atomicAdd(&C[(size_t)row * 64 + col], acc[v]);
    }
}

// out[dst[e]*64 + c] += m[src[e]*64 + c] * attr[e]   (one thread per edge-channel)
__global__ void seg_add_kernel(const float* __restrict__ m,
                               const int* __restrict__ dst,
                               const int* __restrict__ src,
                               const float* __restrict__ attr,
                               float* out, int E)
{
    int idx = blockIdx.x * blockDim.x + threadIdx.x;
    int e = idx >> 6;
    int c = idx & 63;
    if (e >= E) return;
    int s = src[e];
    int d = dst[e];
    float a = attr[e];
    if (c == 0 && e + 64 < E) {
        __builtin_prefetch(&m[(size_t)src[e + 64] * CC], 0, 1);   // global_prefetch_b8
    }
    atomicAdd(&out[(size_t)d * CC + c], m[(size_t)s * CC + c] * a);
}

__global__ void relu_kernel(float* __restrict__ x, int n) {
    int i = blockIdx.x * blockDim.x + threadIdx.x;
    if (i < n) x[i] = fmaxf(x[i], 0.0f);
}

__global__ void zero_kernel(float* __restrict__ x, int n) {
    int i = blockIdx.x * blockDim.x + threadIdx.x;
    if (i < n) x[i] = 0.0f;
}

// Row-wise log_softmax over C=64, one wave32 per row, shuffle reductions.
__global__ __launch_bounds__(32)
void logsoftmax_rows_kernel(const float* __restrict__ in, float* __restrict__ out, int n)
{
    int row  = blockIdx.x;
    int lane = threadIdx.x;
    if (row >= n) return;
    float v0 = in[(size_t)row * CC + lane];
    float v1 = in[(size_t)row * CC + 32 + lane];
    float mx = fmaxf(v0, v1);
#pragma unroll
    for (int off = 16; off > 0; off >>= 1) mx = fmaxf(mx, __shfl_xor(mx, off, 32));
    float s = __expf(v0 - mx) + __expf(v1 - mx);
#pragma unroll
    for (int off = 16; off > 0; off >>= 1) s += __shfl_xor(s, off, 32);
    float lse = mx + __logf(s);
    out[(size_t)row * CC + lane]      = v0 - lse;
    out[(size_t)row * CC + 32 + lane] = v1 - lse;
}

// pooled = mean over rows of hd[64,64]; h1 = leaky_relu(pooled@W1+b1);
// logits = h1@W2+b2; out = log_softmax(logits).  Single block, 64 threads.
__global__ __launch_bounds__(64)
void head_kernel(const float* __restrict__ hd,
                 const float* __restrict__ W1, const float* __restrict__ b1,
                 const float* __restrict__ W2, const float* __restrict__ b2,
                 float* __restrict__ out)
{
    __shared__ float pooled[64];
    __shared__ float h1[32];
    __shared__ float logits[32];
    __shared__ float lse;
    int t = threadIdx.x;
    float sum = 0.0f;
    for (int r = 0; r < 64; ++r) sum += hd[r * 64 + t];
    pooled[t] = sum * (1.0f / 64.0f);
    __syncthreads();
    if (t < 32) {
        float a = b1[t];
        for (int k = 0; k < 64; ++k) a += pooled[k] * W1[k * 32 + t];
        h1[t] = (a > 0.0f) ? a : 0.01f * a;   // jax leaky_relu default slope
    }
    __syncthreads();
    if (t < 32) {
        float a = b2[t];
        for (int k = 0; k < 32; ++k) a += h1[k] * W2[k * 32 + t];
        logits[t] = a;
    }
    __syncthreads();
    if (t == 0) {
        float mx = -1e30f;
        for (int i = 0; i < 32; ++i) mx = fmaxf(mx, logits[i]);
        float s = 0.0f;
        for (int i = 0; i < 32; ++i) s += __expf(logits[i] - mx);
        lse = mx + __logf(s);
    }
    __syncthreads();
    if (t < 32) out[t] = logits[t] - lse;
}

// ---------------------------------------------------------------------------
// Host orchestration
// ---------------------------------------------------------------------------
static inline void fast_gemm(hipStream_t st, const float* A, int lda, const float* B,
                             float* C, int M, int K, const float* bias, int acc, int relu)
{
    wmma_gemm_fast_kernel<<<dim3((M + 63) / 64), 128, 0, st>>>(
        A, lda, B, C, M, K, bias, acc, relu);
}

static inline void tA_gemm(hipStream_t st, const float* P, const float* B, float* C)
{
    zero_kernel<<<(CC * CC + 255) / 256, 256, 0, st>>>(C, CC * CC);
    const int kChunk = 2048;
    const int nch = (NN + kChunk - 1) / kChunk;
    wmma_gemm_tA_kernel<<<dim3(CC / 16, CC / 16, nch), 32, 0, st>>>(P, B, C, NN, kChunk);
}

static inline void seg_add(hipStream_t st, const float* m, const int* edges,
                           const float* attr, float* out)
{
    int nthreads = EE * CC;
    seg_add_kernel<<<(nthreads + 255) / 256, 256, 0, st>>>(
        m, edges, edges + EE, attr, out, EE);
}

static inline void relu_all(hipStream_t st, float* x, int n) {
    relu_kernel<<<(n + 255) / 256, 256, 0, st>>>(x, n);
}

// 2-layer hetero GNN over the edge lists; `out` gets the relu'd layer-2 result.
static void run_sparse_gnn(hipStream_t st, const float* x,
                           const int* e0, const float* a0,
                           const int* e1, const float* a1,
                           const float* Ws0, const float* Wr0,
                           const float* Ws1, const float* Wr1, const float* b,
                           float* h1buf, float* mbuf, float* out)
{
    const int*   es[2] = {e0, e1};
    const float* as[2] = {a0, a1};
    // layer 1 (K = F = 128)
    fast_gemm(st, x, FF, Ws0, h1buf, NN, FF, b, 0, 0);
    for (int r = 0; r < 2; ++r) {
        fast_gemm(st, x, FF, Wr0 + (size_t)r * FF * CC, mbuf, NN, FF, nullptr, 0, 0);
        seg_add(st, mbuf, es[r], as[r], h1buf);
    }
    relu_all(st, h1buf, NN * CC);
    // layer 2 (K = C = 64)
    fast_gemm(st, h1buf, CC, Ws1, out, NN, CC, b + CC, 0, 0);
    for (int r = 0; r < 2; ++r) {
        fast_gemm(st, h1buf, CC, Wr1 + (size_t)r * CC * CC, mbuf, NN, CC, nullptr, 0, 0);
        seg_add(st, mbuf, es[r], as[r], out);
    }
    relu_all(st, out, NN * CC);
}

extern "C" void kernel_launch(void* const* d_in, const int* in_sizes, int n_in,
                              void* d_out, int out_size, void* d_ws, size_t ws_size,
                              hipStream_t stream)
{
    (void)in_sizes; (void)n_in; (void)out_size; (void)ws_size;
    const float* x   = (const float*)d_in[0];
    const int*   e0  = (const int*)d_in[1];
    const int*   e1  = (const int*)d_in[2];
    const float* a0  = (const float*)d_in[3];
    const float* a1  = (const float*)d_in[4];
    const float* g1p_Ws0 = (const float*)d_in[5];
    const float* g1p_Wr0 = (const float*)d_in[6];
    const float* g1p_Ws1 = (const float*)d_in[7];
    const float* g1p_Wr1 = (const float*)d_in[8];
    const float* g1p_b   = (const float*)d_in[9];
    const float* g1e_Ws0 = (const float*)d_in[10];
    const float* g1e_Wr0 = (const float*)d_in[11];
    const float* g1e_Ws1 = (const float*)d_in[12];
    const float* g1e_Wr1 = (const float*)d_in[13];
    const float* g1e_b   = (const float*)d_in[14];
    const float* g2_Ws0  = (const float*)d_in[15];
    const float* g2_Wr0  = (const float*)d_in[16];
    const float* g2_Ws1  = (const float*)d_in[17];
    const float* g2_Wr1  = (const float*)d_in[18];
    const float* g2_b    = (const float*)d_in[19];
    const float* lin1_W  = (const float*)d_in[20];
    const float* lin1_b  = (const float*)d_in[21];
    const float* lin2_W  = (const float*)d_in[22];
    const float* lin2_b  = (const float*)d_in[23];

    // Output layout: [log_softmax(logits) 32] [s N*64] [x1 64*64]
    float* out    = (float*)d_out;
    float* out_s  = out + 32;
    float* out_x1 = out + 32 + (size_t)NN * CC;

    // Workspace layout
    float* w     = (float*)d_ws;
    float* h1buf = w;                          // N*C
    float* mbuf  = w + (size_t)NN * CC;        // N*C
    float* xebuf = w + 2 * (size_t)NN * CC;    // N*C
    float* Pbuf  = w + 3 * (size_t)NN * CC;    // N*C
    float* A0    = w + 4 * (size_t)NN * CC;    // C*C
    float* A1    = A0 + CC * CC;
    float* tbuf  = A1 + CC * CC;
    float* hd    = tbuf + CC * CC;
    float* hd2   = hd + CC * CC;

    // --- two sparse hetero GNNs ---
    run_sparse_gnn(stream, x, e0, a0, e1, a1,
                   g1p_Ws0, g1p_Wr0, g1p_Ws1, g1p_Wr1, g1p_b,
                   h1buf, mbuf, out_s);                       // s -> d_out
    run_sparse_gnn(stream, x, e0, a0, e1, a1,
                   g1e_Ws0, g1e_Wr0, g1e_Ws1, g1e_Wr1, g1e_b,
                   h1buf, mbuf, xebuf);                       // xe

    // --- assignment P = log_softmax(s, axis=1) ---
    logsoftmax_rows_kernel<<<NN, 32, 0, stream>>>(out_s, Pbuf, NN);

    // --- coarsen: A_r = P^T @ segment_sum(P[src]*attr -> dst); x1 = P^T @ xe ---
    const int*   es[2] = {e0, e1};
    const float* as[2] = {a0, a1};
    float*       Ar[2] = {A0, A1};
    for (int r = 0; r < 2; ++r) {
        zero_kernel<<<(NN * CC + 255) / 256, 256, 0, stream>>>(mbuf, NN * CC);
        seg_add(stream, Pbuf, es[r], as[r], mbuf);
        tA_gemm(stream, Pbuf, mbuf, Ar[r]);                   // A_r = P^T @ M_r
    }
    tA_gemm(stream, Pbuf, xebuf, out_x1);                     // x1 = P^T @ xe -> d_out

    // --- dense hetero GNN on [64,64] ---
    // layer 1
    fast_gemm(stream, out_x1, CC, g2_Ws0, hd, CC, CC, g2_b, 0, 0);
    for (int r = 0; r < 2; ++r) {
        fast_gemm(stream, out_x1, CC, g2_Wr0 + (size_t)r * CC * CC, tbuf, CC, CC, nullptr, 0, 0);
        fast_gemm(stream, Ar[r], CC, tbuf, hd, CC, CC, nullptr, 1, 0);   // hd += A_r @ t
    }
    relu_all(stream, hd, CC * CC);
    // layer 2
    fast_gemm(stream, hd, CC, g2_Ws1, hd2, CC, CC, g2_b + CC, 0, 0);
    for (int r = 0; r < 2; ++r) {
        fast_gemm(stream, hd, CC, g2_Wr1 + (size_t)r * CC * CC, tbuf, CC, CC, nullptr, 0, 0);
        fast_gemm(stream, Ar[r], CC, tbuf, hd2, CC, CC, nullptr, 1, 0);  // hd2 += A_r @ t
    }
    relu_all(stream, hd2, CC * CC);

    // --- pooled head + final log_softmax ---
    head_kernel<<<1, 64, 0, stream>>>(hd2, lin1_W, lin1_b, lin2_W, lin2_b, out);
}
